// CPI_MPNN_3255585210687
// MI455X (gfx1250) — matrix operand order repaired and checked by reading, not compiled, not run on hardware
//
#include <hip/hip_runtime.h>
#include <hip/hip_bf16.h>

typedef __attribute__((ext_vector_type(16))) _Float16 v16h;
typedef __attribute__((ext_vector_type(8)))  _Float16 v8h;
typedef __attribute__((ext_vector_type(8)))  float    v8f;

// ---------------- problem constants ----------------
#define NMOL 128
#define NATM 48
#define NBND 96
#define ATOMF 39
#define BONDF 50        // ATOM_FDIM + BOND_FDIM = 39+11
#define HLD  224        // hidden (200) padded to 7*32 for WMMA K and N
#define LSEQ 1000
#define LPAD 1006       // 3 halo rows each side
#define PCHUNK 32       // proteins per tower pass

// =====================================================================
// Generic wave32 WMMA GEMM: each wave computes a 32(M) x 32(N) tile of
//   D = A(MxK,f16,row-major) * B(KxN, packed N-major f16) [+bias][+addin]
// 4 v_wmma per k-step with A reused across N-tiles and B across M-tiles
// (2 b128 loads per WMMA). Optional relu'd f16 store, optional f32
// store, optional fused column-max (atomicMax on non-negative floats).
// grid = (ceil(M/32), Npad/32, batch), block = 32 (one wave).
// =====================================================================
__global__ __launch_bounds__(32) void gemm_wmma_f16(
    const _Float16* __restrict__ A, long long aBatch, int lda,
    const _Float16* __restrict__ Bw, int Kpad,
    const float* __restrict__ bias,
    const float* __restrict__ addin, long long addBatch, int ldadd,
    _Float16* __restrict__ Ch, long long cBatch, int ldc,
    float* __restrict__ Cf, long long cfBatch, int ldcf,
    float* __restrict__ maxOut, int maxLd,
    int M, int reluH, int reluF)
{
    const int lane = threadIdx.x;
    const int hs   = lane >> 4;        // lane half (ISA 7.12.2 operand layouts)
    const int mr   = lane & 15;
    const int z    = blockIdx.z;
    const int row0 = blockIdx.x * 32;
    const int n0   = blockIdx.y * 32;
    // A-matrix K split per lane half: {0..7,16..23} / {8..15,24..31}
    const int klo = hs ? 8  : 0;
    const int khi = hs ? 24 : 16;
    // B-matrix: same K split per lane half as A

    const _Float16* Ab = A + (long long)z * aBatch;
    int ar0 = row0 + mr;      if (ar0 > M - 1) ar0 = M - 1;   // clamp edges
    int ar1 = row0 + 16 + mr; if (ar1 > M - 1) ar1 = M - 1;
    const _Float16* Ap0 = Ab + (long long)ar0 * lda;
    const _Float16* Ap1 = Ab + (long long)ar1 * lda;
    const _Float16* B0p = Bw + (long long)(n0 + mr) * Kpad;
    const _Float16* B1p = B0p + (long long)16 * Kpad;

    v8f acc00 = {0.f,0.f,0.f,0.f,0.f,0.f,0.f,0.f};
    v8f acc01 = acc00, acc10 = acc00, acc11 = acc00;

    for (int k0 = 0; k0 < Kpad; k0 += 32) {
        v8h a0lo = *(const v8h*)(Ap0 + k0 + klo);
        v8h a0hi = *(const v8h*)(Ap0 + k0 + khi);
        v8h a1lo = *(const v8h*)(Ap1 + k0 + klo);
        v8h a1hi = *(const v8h*)(Ap1 + k0 + khi);
        v8h b0lo = *(const v8h*)(B0p + k0 + klo);
        v8h b0hi = *(const v8h*)(B0p + k0 + khi);
        v8h b1lo = *(const v8h*)(B1p + k0 + klo);
        v8h b1hi = *(const v8h*)(B1p + k0 + khi);
        v16h a0, a1, b0, b1;
#pragma unroll
        for (int e = 0; e < 8; ++e) {
            a0[e] = a0lo[e];  a0[e + 8] = a0hi[e];
            a1[e] = a1lo[e];  a1[e + 8] = a1hi[e];
            b0[e] = b0lo[e];  b0[e + 8] = b0hi[e];
            b1[e] = b1lo[e];  b1[e + 8] = b1hi[e];
        }
        acc00 = __builtin_amdgcn_wmma_f32_16x16x32_f16(false, a0, false, b0,
                                                       (short)0, acc00, false, false);
        acc01 = __builtin_amdgcn_wmma_f32_16x16x32_f16(false, a0, false, b1,
                                                       (short)0, acc01, false, false);
        acc10 = __builtin_amdgcn_wmma_f32_16x16x32_f16(false, a1, false, b0,
                                                       (short)0, acc10, false, false);
        acc11 = __builtin_amdgcn_wmma_f32_16x16x32_f16(false, a1, false, b1,
                                                       (short)0, acc11, false, false);
    }

    const int hs8 = hs ? 8 : 0;        // D layout: rows r / r+8 by lane half
    float localMax0 = 0.f, localMax1 = 0.f;

    auto emit = [&](const v8f& acc, int mt, int nt) {
        const int n  = n0 + mr + nt * 16;
        const float bv = bias ? bias[n] : 0.0f;
        const int mb = row0 + mt * 16 + hs8;
        float lm = 0.f;
        for (int r = 0; r < 8; ++r) {
            const int m = mb + r;
            if (m >= M) break;
            float v = acc[r] + bv;
            if (addin) v += addin[(long long)z * addBatch + (long long)m * ldadd + n];
            float vH = (reluH && v < 0.f) ? 0.f : v;
            if (Ch) Ch[(long long)z * cBatch + (long long)m * ldc + n] = (_Float16)vH;
            if (Cf) {
                float vF = (reluF && v < 0.f) ? 0.f : v;
                Cf[(long long)z * cfBatch + (long long)m * ldcf + n] = vF;
            }
            if (vH > lm) lm = vH;
        }
        if (nt == 0) { if (lm > localMax0) localMax0 = lm; }
        else         { if (lm > localMax1) localMax1 = lm; }
    };
    emit(acc00, 0, 0); emit(acc10, 1, 0);
    emit(acc01, 0, 1); emit(acc11, 1, 1);

    if (maxOut) {      // post-relu values >= 0: uint compare == float compare
        atomicMax((unsigned int*)(maxOut + (long long)z * maxLd + n0 + mr),
                  __float_as_uint(localMax0));
        atomicMax((unsigned int*)(maxOut + (long long)z * maxLd + n0 + mr + 16),
                  __float_as_uint(localMax1));
    }
}

// ------------------- packing kernels (run every launch) -------------------
// (in,out)-layout f32 weight -> N-major f16, zero padded
__global__ void k_pack_nk(const float* __restrict__ W, int K, int N,
                          int Kpad, int Npad, _Float16* __restrict__ out)
{
    int idx = blockIdx.x * 256 + threadIdx.x;
    int total = Npad * Kpad;
    if (idx >= total) return;
    int n = idx / Kpad, k = idx % Kpad;
    float v = (k < K && n < N) ? W[k * N + n] : 0.f;
    out[idx] = (_Float16)v;
}

// conv weight [co][ci][t] -> out[co][t*Cis+ci] (N-major implicit-GEMM layout)
__global__ void k_pack_conv(const float* __restrict__ W, int Ci, int kw, int Cis,
                            int N, int Npad, _Float16* __restrict__ out)
{
    int Kpad = kw * Cis;
    int idx = blockIdx.x * 256 + threadIdx.x;
    int total = Npad * Kpad;
    if (idx >= total) return;
    int n = idx / Kpad, r = idx % Kpad;
    int t = r / Cis, ci = r % Cis;
    float v = (n < N && ci < Ci) ? W[(n * Ci + ci) * kw + t] : 0.f;
    out[idx] = (_Float16)v;
}

__global__ void k_pack_bias(const float* __restrict__ b, int N, int Npad,
                            float* __restrict__ out)
{
    int idx = blockIdx.x * 256 + threadIdx.x;
    if (idx < Npad) out[idx] = (idx < N) ? b[idx] : 0.f;
}

// fbonds f32 (rows x 50) -> f16 (rows x 64 zero padded)
__global__ void k_pack_rows(const float* __restrict__ X, int K, int Kpad,
                            long long total, _Float16* __restrict__ out)
{
    long long idx = (long long)blockIdx.x * 256 + threadIdx.x;
    if (idx >= total) return;
    int k = (int)(idx % Kpad);
    long long r = idx / Kpad;
    out[idx] = (_Float16)((k < K) ? X[r * K + k] : 0.f);
}

// ------------------- MPNN gathers -------------------
__global__ void k_gather_bonds(const _Float16* __restrict__ msg,
                               const int* __restrict__ bg,
                               _Float16* __restrict__ nei)
{
    int i = blockIdx.x, b = blockIdx.y, k = threadIdx.x;   // 224 threads
    const int* g = bg + ((long long)b * NBND + i) * 6;
    float s = 0.f;
#pragma unroll
    for (int j = 0; j < 6; ++j)
        s += (float)msg[((long long)b * NBND + g[j]) * HLD + k];
    nei[((long long)b * NBND + i) * HLD + k] = (_Float16)s;
}

// atom gather + [fatoms | nei] concat, K padded to 256
__global__ void k_gather_atoms(const _Float16* __restrict__ msg,
                               const int* __restrict__ ag,
                               const float* __restrict__ fatoms,
                               _Float16* __restrict__ Ao)
{
    int i = blockIdx.x, b = blockIdx.y, k = threadIdx.x;   // 256 threads
    float v = 0.f;
    if (k < ATOMF) {
        v = fatoms[((long long)b * NATM + i) * ATOMF + k];
    } else if (k < ATOMF + 200) {
        const int* g = ag + ((long long)b * NATM + i) * 6;
        float s = 0.f;
#pragma unroll
        for (int j = 0; j < 6; ++j)
            s += (float)msg[((long long)b * NBND + g[j]) * HLD + (k - ATOMF)];
        v = s;
    }
    Ao[((long long)b * NATM + i) * 256 + k] = (_Float16)v;
}

// mean over atoms -> fc input cols [0,200)
__global__ void k_mean(const float* __restrict__ atomF, _Float16* __restrict__ fcin)
{
    int b = blockIdx.x, h = threadIdx.x;
    if (h >= 200) return;
    float s = 0.f;
    for (int i = 0; i < NATM; ++i)
        s += atomF[((long long)b * NATM + i) * HLD + h];
    fcin[(long long)b * 416 + h] = (_Float16)(s * (1.f / 48.f));
}

// ------------------- protein tower helpers -------------------
__global__ void k_embed(const int* __restrict__ seq, const float* __restrict__ emb,
                        _Float16* __restrict__ pv0, int b0)
{
    int r = blockIdx.x, b = blockIdx.y, ci = threadIdx.x;  // 64 threads
    float v = 0.f;
    if (r >= 3 && r < 3 + LSEQ && ci < 50) {
        int a = seq[(long long)(b0 + b) * LSEQ + (r - 3)];
        v = emb[a * 50 + ci];
    }
    pv0[((long long)b * LPAD + r) * 64 + ci] = (_Float16)v;
}

__global__ void k_zero_halos(_Float16* __restrict__ pv1, _Float16* __restrict__ pv2)
{
    int hi = blockIdx.x, b = blockIdx.y, c = threadIdx.x;  // 128 threads
    int r = (hi < 3) ? hi : (1000 + hi);                   // 0,1,2,1003,1004,1005
    if (c < 96) pv1[((long long)b * LPAD + r) * 96 + c] = (_Float16)0.f;
    pv2[((long long)b * LPAD + r) * 128 + c] = (_Float16)0.f;
}

__global__ void k_init(float* __restrict__ maxbuf, _Float16* __restrict__ fcin)
{
    int idx = blockIdx.x * 256 + threadIdx.x;
    const int nMax = NMOL * HLD;
    if (idx < nMax) { maxbuf[idx] = 0.f; return; }
    idx -= nMax;
    if (idx < NMOL * 16) {                 // zero fc-input K tail cols 400..415
        int b = idx / 16, c = idx % 16;
        fcin[(long long)b * 416 + 400 + c] = (_Float16)0.f;
    }
}

__global__ void k_fcin_prot(const float* __restrict__ maxbuf, _Float16* __restrict__ fcin)
{
    int b = blockIdx.x, c = threadIdx.x;
    if (c < 200) fcin[(long long)b * 416 + 200 + c] = (_Float16)maxbuf[(long long)b * HLD + c];
}

__global__ void k_fc2(const _Float16* __restrict__ h2, const float* __restrict__ w,
                      const float* __restrict__ bb, float* __restrict__ out)
{
    int b = blockIdx.x * blockDim.x + threadIdx.x;
    if (b >= NMOL) return;
    float s = bb[0];
    for (int k = 0; k < 100; ++k) s += (float)h2[(long long)b * 128 + k] * w[k];
    out[b] = s;
}

// ------------------- host-side launch helper -------------------
static inline void launch_gemm(hipStream_t s, dim3 g,
    const _Float16* A, long long aB, int lda,
    const _Float16* Bw, int Kpad, const float* bias,
    const float* addin, long long addB, int ldadd,
    _Float16* Ch, long long cB, int ldc,
    float* Cf, long long cfB, int ldcf,
    float* maxOut, int maxLd, int M, int reluH, int reluF)
{
    gemm_wmma_f16<<<g, 32, 0, s>>>(A, aB, lda, Bw, Kpad, bias, addin, addB, ldadd,
                                   Ch, cB, ldc, Cf, cfB, ldcf, maxOut, maxLd,
                                   M, reluH, reluF);
}

extern "C" void kernel_launch(void* const* d_in, const int* in_sizes, int n_in,
                              void* d_out, int out_size, void* d_ws, size_t ws_size,
                              hipStream_t stream)
{
    (void)in_sizes; (void)n_in; (void)out_size; (void)ws_size;
    const float* fatoms = (const float*)d_in[0];
    const float* fbonds = (const float*)d_in[1];
    const int*   agraph = (const int*)d_in[2];
    const int*   bgraph = (const int*)d_in[3];
    const int*   pseq   = (const int*)d_in[4];
    const float* W_i    = (const float*)d_in[5];
    const float* W_h    = (const float*)d_in[6];
    const float* W_o_w  = (const float*)d_in[7];
    const float* W_o_b  = (const float*)d_in[8];
    const float* embp   = (const float*)d_in[9];
    const float* c0w = (const float*)d_in[10]; const float* c0b = (const float*)d_in[11];
    const float* c1w = (const float*)d_in[12]; const float* c1b = (const float*)d_in[13];
    const float* c2w = (const float*)d_in[14]; const float* c2b = (const float*)d_in[15];
    const float* f0w = (const float*)d_in[16]; const float* f0b = (const float*)d_in[17];
    const float* f1w = (const float*)d_in[18]; const float* f1b = (const float*)d_in[19];
    const float* f2w = (const float*)d_in[20]; const float* f2b = (const float*)d_in[21];

    // ---- workspace carve-up (all L2-resident; ~52 MB) ----
    size_t off = 0;
    auto alloc = [&](size_t elems, size_t esz) -> char* {
        char* p = (char*)d_ws + off;
        off = (off + elems * esz + 255) & ~(size_t)255;
        return p;
    };
    _Float16* WiH  = (_Float16*)alloc(224 * 64, 2);
    _Float16* WhH  = (_Float16*)alloc(224 * 224, 2);
    _Float16* WoH  = (_Float16*)alloc(224 * 256, 2);
    _Float16* B0   = (_Float16*)alloc(96  * 192, 2);
    _Float16* B1   = (_Float16*)alloc(128 * 480, 2);
    _Float16* B2   = (_Float16*)alloc(224 * 896, 2);
    _Float16* F0   = (_Float16*)alloc(224 * 416, 2);
    _Float16* F1   = (_Float16*)alloc(128 * 224, 2);
    float* bWo = (float*)alloc(224, 4);
    float* bC0 = (float*)alloc(96, 4);
    float* bC1 = (float*)alloc(128, 4);
    float* bC2 = (float*)alloc(224, 4);
    float* bF0 = (float*)alloc(224, 4);
    float* bF1 = (float*)alloc(128, 4);
    _Float16* fbH    = (_Float16*)alloc((size_t)NMOL * NBND * 64, 2);
    float*    binput = (float*)   alloc((size_t)NMOL * NBND * HLD, 4);
    _Float16* msg    = (_Float16*)alloc((size_t)NMOL * NBND * HLD, 2);
    _Float16* nei    = (_Float16*)alloc((size_t)NMOL * NBND * HLD, 2);
    _Float16* AoH    = (_Float16*)alloc((size_t)NMOL * NATM * 256, 2);
    float*    atomF  = (float*)   alloc((size_t)NMOL * NATM * HLD, 4);
    _Float16* fcin   = (_Float16*)alloc((size_t)NMOL * 416, 2);
    _Float16* h1     = (_Float16*)alloc((size_t)NMOL * 224, 2);
    _Float16* h2     = (_Float16*)alloc((size_t)NMOL * 128, 2);
    float*    maxbuf = (float*)   alloc((size_t)NMOL * HLD, 4);
    _Float16* pv0    = (_Float16*)alloc((size_t)PCHUNK * LPAD * 64, 2);
    _Float16* pv1    = (_Float16*)alloc((size_t)PCHUNK * LPAD * 96, 2);
    _Float16* pv2    = (_Float16*)alloc((size_t)PCHUNK * LPAD * 128, 2);

    // ---- weight packing (cheap; every launch for determinism) ----
    auto blk = [](int n) { return (n + 255) / 256; };
    k_pack_nk<<<blk(224*64),  256, 0, stream>>>(W_i,  BONDF, 200, 64,  224, WiH);
    k_pack_nk<<<blk(224*224), 256, 0, stream>>>(W_h,  200,   200, 224, 224, WhH);
    k_pack_nk<<<blk(224*256), 256, 0, stream>>>(W_o_w, ATOMF+200, 200, 256, 224, WoH);
    k_pack_nk<<<blk(224*416), 256, 0, stream>>>(f0w,  400,   200, 416, 224, F0);
    k_pack_nk<<<blk(128*224), 256, 0, stream>>>(f1w,  200,   100, 224, 128, F1);
    k_pack_conv<<<blk(96*192),  256, 0, stream>>>(c0w, 50,  3, 64,  96,  96,  B0);
    k_pack_conv<<<blk(128*480), 256, 0, stream>>>(c1w, 96,  5, 96,  128, 128, B1);
    k_pack_conv<<<blk(224*896), 256, 0, stream>>>(c2w, 128, 7, 128, 200, 224, B2);
    k_pack_bias<<<1, 256, 0, stream>>>(W_o_b, 200, 224, bWo);
    k_pack_bias<<<1, 256, 0, stream>>>(c0b, 96,  96,  bC0);
    k_pack_bias<<<1, 256, 0, stream>>>(c1b, 128, 128, bC1);
    k_pack_bias<<<1, 256, 0, stream>>>(c2b, 200, 224, bC2);
    k_pack_bias<<<1, 256, 0, stream>>>(f0b, 200, 224, bF0);
    k_pack_bias<<<1, 256, 0, stream>>>(f1b, 100, 128, bF1);

    k_init<<<blk(NMOL*HLD + NMOL*16), 256, 0, stream>>>(maxbuf, fcin);

    // ---- MPNN ----
    long long nbRows = (long long)NMOL * NBND;           // 12288
    k_pack_rows<<<blk((int)(nbRows*64)), 256, 0, stream>>>(fbonds, BONDF, 64, nbRows*64, fbH);

    // binput = fb @ W_i  (f32 kept pre-relu), msg = relu(binput) f16
    launch_gemm(stream, dim3((unsigned)(nbRows/32), 224/32, 1),
                fbH, 0, 64, WiH, 64, nullptr, nullptr, 0, 0,
                msg, 0, HLD, binput, 0, HLD, nullptr, 0, (int)nbRows, 1, 0);

    for (int d = 0; d < 2; ++d) {   // DEPTH-1 message-passing steps
        k_gather_bonds<<<dim3(NBND, NMOL), HLD, 0, stream>>>(msg, bgraph, nei);
        launch_gemm(stream, dim3((unsigned)(nbRows/32), 224/32, 1),
                    nei, 0, HLD, WhH, HLD, nullptr, binput, 0, HLD,
                    msg, 0, HLD, nullptr, 0, 0, nullptr, 0, (int)nbRows, 1, 0);
    }
    k_gather_atoms<<<dim3(NATM, NMOL), 256, 0, stream>>>(msg, agraph, fatoms, AoH);
    long long naRows = (long long)NMOL * NATM;           // 6144
    launch_gemm(stream, dim3((unsigned)(naRows/32), 224/32, 1),
                AoH, 0, 256, WoH, 256, bWo, nullptr, 0, 0,
                nullptr, 0, 0, atomF, 0, HLD, nullptr, 0, (int)naRows, 0, 1);
    k_mean<<<NMOL, 256, 0, stream>>>(atomF, fcin);

    // ---- protein CNN tower (implicit GEMM, 4 chunks of 32) ----
    const int tilesM = (LSEQ + 31) / 32;   // 32 (rows >= 1000 clamped/masked)
    for (int c = 0; c < NMOL / PCHUNK; ++c) {
        int b0 = c * PCHUNK;
        k_embed<<<dim3(LPAD, PCHUNK), 64, 0, stream>>>(pseq, embp, pv0, b0);
        k_zero_halos<<<dim3(6, PCHUNK), 128, 0, stream>>>(pv1, pv2);
        // conv0: Ci=50(->64) k=3 off=1 : K=192, N=96
        launch_gemm(stream, dim3(tilesM, 96/32, PCHUNK),
                    pv0 + 2*64, (long long)LPAD*64, 64, B0, 192, bC0,
                    nullptr, 0, 0, pv1 + 3*96, (long long)LPAD*96, 96,
                    nullptr, 0, 0, nullptr, 0, LSEQ, 1, 0);
        // conv1: Ci=96 k=5 off=2 : K=480, N=128
        launch_gemm(stream, dim3(tilesM, 128/32, PCHUNK),
                    pv1 + 1*96, (long long)LPAD*96, 96, B1, 480, bC1,
                    nullptr, 0, 0, pv2 + 3*128, (long long)LPAD*128, 128,
                    nullptr, 0, 0, nullptr, 0, LSEQ, 1, 0);
        // conv2: Ci=128 k=7 off=3 : K=896, N=224 ; fused relu + max over L
        launch_gemm(stream, dim3(tilesM, 224/32, PCHUNK),
                    pv2, (long long)LPAD*128, 128, B2, 896, bC2,
                    nullptr, 0, 0, nullptr, 0, 0, nullptr, 0, 0,
                    maxbuf + (long long)b0*HLD, HLD, LSEQ, 1, 0);
    }
    k_fcin_prot<<<NMOL, 256, 0, stream>>>(maxbuf, fcin);

    // ---- FC head ----
    launch_gemm(stream, dim3(NMOL/32, 224/32, 1),
                fcin, 0, 416, F0, 416, bF0, nullptr, 0, 0,
                h1, 0, 224, nullptr, 0, 0, nullptr, 0, NMOL, 1, 0);
    launch_gemm(stream, dim3(NMOL/32, 128/32, 1),
                h1, 0, 224, F1, 224, bF1, nullptr, 0, 0,
                h2, 0, 128, nullptr, 0, 0, nullptr, 0, NMOL, 1, 0);
    k_fc2<<<1, 128, 0, stream>>>(h2, f2w, f2b, (float*)d_out);
}
